// BiAttention_18502719111880
// MI455X (gfx1250) — matrix operand order repaired and checked by reading, not compiled, not run on hardware
//
#include <hip/hip_runtime.h>
#include <hip/hip_bf16.h>

// ---------------- problem constants ----------------
constexpr int CB  = 16;    // batch
constexpr int CLD = 4096;  // Ld
constexpr int CLM = 1024;  // Lm
constexpr int CH  = 128;   // H

typedef __attribute__((ext_vector_type(16))) _Float16 v16h;
typedef __attribute__((ext_vector_type(8)))  float    v8f;

// ---------------- WMMA fragment helpers (wave32, 16x16x32 f16) ----------------
// A-matrix 16x32 (MxK), row-major source with leading dim ld:
// lane<16 : row=lane,   k = 0..15
// lane>=16: row=lane-16, k = 16..31
__device__ __forceinline__ v16h load_A16x32(const _Float16* __restrict__ src, int ld) {
  int lane = threadIdx.x & 31;
  int half = lane >> 4, row = lane & 15;
  const _Float16* p = src + (size_t)row * ld + half * 16;
  v16h a;
#pragma unroll
  for (int j = 0; j < 16; ++j) a[j] = p[j];
  return a;
}

// B-matrix 32x16 (KxN) where B = W^T for row-major W[N][K] (caller passes W + n0*ld + k0):
// lane<16 : col=lane,    k = 0..15  (16 contiguous f16 = one 32B load)
// lane>=16: col=lane-16, k = 16..31
__device__ __forceinline__ v16h load_BT32x16(const _Float16* __restrict__ W, int ld) {
  int lane = threadIdx.x & 31;
  int half = lane >> 4, col = lane & 15;
  const _Float16* p = W + (size_t)col * ld + half * 16;
  v16h b;
#pragma unroll
  for (int j = 0; j < 16; ++j) b[j] = p[j];
  return b;
}

// B-fragment read where the caller already resolved (col,half) row addressing (LDS tiles).
__device__ __forceinline__ v16h load_B_lds(const _Float16* p) {
  v16h b;
#pragma unroll
  for (int j = 0; j < 16; ++j) b[j] = p[j];
  return b;
}

__device__ __forceinline__ v8f wmma_f16(v16h a, v16h b, v8f c) {
  return __builtin_amdgcn_wmma_f32_16x16x32_f16(false, a, false, b, (short)0, c, false, false);
}

// C/D layout: VGPR j: lanes 0-15 -> (row j, col lane); lanes 16-31 -> (row j+8, col lane-16)

// ---------------- async global->LDS helpers (CDNA5, ASYNCcnt-tracked) ----------------
__device__ __forceinline__ uint32_t lds_off(const void* p) {
  // generic -> LDS address space; ptrtoint of AS(3) pointer yields workgroup-LDS byte offset
  return (uint32_t)(uintptr_t)(__attribute__((address_space(3))) const void*)p;
}

__device__ __forceinline__ void async_g2l_b128(uint32_t lds_byte, uint64_t gaddr) {
  asm volatile("global_load_async_to_lds_b128 %0, %1, off"
               :: "v"(lds_byte), "v"(gaddr)
               : "memory");
}

__device__ __forceinline__ void wait_async0() {
  asm volatile("s_wait_asynccnt 0" ::: "memory");
}

// ---------------- prep kernels ----------------
__global__ __launch_bounds__(128)
void prep_x_kernel(const float* __restrict__ x, const float* __restrict__ w_i1,
                   _Float16* __restrict__ xh, float* __restrict__ input_dot) {
  __shared__ float red[128];
  size_t row = blockIdx.x;  // b*CLD + i
  int t = threadIdx.x;
  float v = x[row * CH + t];
  xh[row * CH + t] = (_Float16)(v * 0.08838834764831845f);  // fold 1/sqrt(H) into A
  red[t] = v * w_i1[t];
  __syncthreads();
  for (int s = 64; s > 0; s >>= 1) { if (t < s) red[t] += red[t + s]; __syncthreads(); }
  if (t == 0) input_dot[row] = red[0];
}

__global__ __launch_bounds__(128)
void prep_mem_kernel(const float* __restrict__ mem, const float* __restrict__ w_m1,
                     const float* __restrict__ W_m2, const float* __restrict__ b_m2,
                     _Float16* __restrict__ memh, _Float16* __restrict__ mem2T,
                     float* __restrict__ memory_dot) {
  __shared__ float rowbuf[128];
  __shared__ float red[128];
  size_t row = blockIdx.x;  // b*CLM + m
  int b = (int)(row / CLM), m = (int)(row % CLM);
  int t = threadIdx.x;
  float v = mem[row * CH + t];
  rowbuf[t] = v;
  memh[row * CH + t] = (_Float16)v;
  red[t] = v * w_m1[t];
  __syncthreads();
  // mem2[m][h=t] = dot(mem_row, W_m2[t,:]) + b_m2[t]; stored transposed [H][Lm]
  float acc = b_m2[t];
#pragma unroll 8
  for (int d = 0; d < CH; ++d) acc = fmaf(rowbuf[d], W_m2[t * CH + d], acc);
  mem2T[((size_t)b * CH + t) * CLM + m] = (_Float16)acc;
  for (int s = 64; s > 0; s >>= 1) { if (t < s) red[t] += red[t + s]; __syncthreads(); }
  if (t == 0) memory_dot[row] = red[0];
}

__global__ __launch_bounds__(256)
void prep_w_kernel(const float* __restrict__ W_i2, const float* __restrict__ W_out,
                   _Float16* __restrict__ Wi2h, _Float16* __restrict__ Wch) {
  int idx = blockIdx.x * blockDim.x + threadIdx.x;
  if (idx < CH * CH)
    Wi2h[idx] = (_Float16)(W_i2[idx] * 11.313708498984761f);  // * sqrt(H), compensates scaled x
  if (idx < 4 * CH * CH) {
    int t = idx / (CH * CH);
    int h = (idx / CH) % CH;
    int k = idx % CH;
    Wch[idx] = (_Float16)(W_out[h * (4 * CH) + t * CH + k]);  // Wc[t][h][k]
  }
}

// ---------------- fused flash-attention + inp2 kernel ----------------
// grid (B, LD/128), 256 threads = 8 waves; each wave owns a 16-row strip.
// mem / mem2T chunks are staged into LDS once per block via async copies (double buffered).
__global__ __launch_bounds__(256)
void attn_kernel(const _Float16* __restrict__ xh, const _Float16* __restrict__ memh,
                 const _Float16* __restrict__ mem2T, const _Float16* __restrict__ Wi2h,
                 const float* __restrict__ b_i2, const float* __restrict__ input_dot,
                 const float* __restrict__ memory_dot,
                 _Float16* __restrict__ inp2h, _Float16* __restrict__ o1h,
                 float* __restrict__ rowmax) {
  __shared__ _Float16 memS[2][32][136];    // mem rows m0..m0+31 (padded +16B/row)
  __shared__ _Float16 mem2S[2][128][40];   // mem2T rows h=0..127, cols m0..m0+31 (padded)
  __shared__ _Float16 Pbuf[8][16][36];     // per-wave 16x32 P tile (padded)

  int wave = threadIdx.x >> 5;
  int lane = threadIdx.x & 31;
  int half = lane >> 4, ln = lane & 15;
  int tt = threadIdx.x;
  int b = blockIdx.x;
  int r0 = blockIdx.y * 128 + wave * 16;

  const _Float16* xrow   = xh + ((size_t)b * CLD + r0) * CH;
  const _Float16* mbase  = memh  + (size_t)b * CLM * CH;
  const _Float16* m2base = mem2T + (size_t)b * CH * CLM;
  const float*    mdotb  = memory_dot + (size_t)b * CLM;

  // async-stage one 32-column chunk into buffer nb (512+512 b128 segments over 256 threads)
  auto issue_chunk = [&](int nb, int mnext) {
#pragma unroll
    for (int q = 0; q < 2; ++q) {
      int s = tt * 2 + q;                      // 0..511
      int row = s >> 4, seg = s & 15;          // memS: 32 rows x 16 segs of 16B
      async_g2l_b128(lds_off(&memS[nb][row][seg * 8]),
                     (uint64_t)(uintptr_t)(mbase + (size_t)(mnext + row) * CH + seg * 8));
    }
#pragma unroll
    for (int q = 0; q < 2; ++q) {
      int s = tt * 2 + q;
      int row = s >> 2, seg = s & 3;           // mem2S: 128 rows x 4 segs of 16B
      async_g2l_b128(lds_off(&mem2S[nb][row][seg * 8]),
                     (uint64_t)(uintptr_t)(m2base + (size_t)row * CLM + mnext + seg * 8));
    }
  };

  issue_chunk(0, 0);  // prefetch chunk 0

  v16h xa[4];
#pragma unroll
  for (int c = 0; c < 4; ++c) xa[c] = load_A16x32(xrow + c * 32, CH);

  // ---- inp2 strip = x_scaled @ (sqrt(H)*W_i2)^T + b_i2 (overlaps with chunk-0 async copy)
#pragma unroll
  for (int nt = 0; nt < 8; ++nt) {
    v8f acc = {};
#pragma unroll
    for (int c = 0; c < 4; ++c)
      acc = wmma_f16(xa[c], load_BT32x16(Wi2h + (size_t)(nt * 16) * CH + c * 32, CH), acc);
    int colg = nt * 16 + ln;
    float bias = b_i2[colg];
#pragma unroll
    for (int j = 0; j < 8; ++j) {
      int rg = r0 + j + half * 8;
      inp2h[((size_t)b * CLD + rg) * CH + colg] = (_Float16)(acc[j] + bias);
    }
  }

  // ---- flash attention over Lm in 32-column chunks
  float mstat[8], lstat[8];
#pragma unroll
  for (int j = 0; j < 8; ++j) { mstat[j] = -3.0e38f; lstat[j] = 0.f; }
  v8f O[8];
#pragma unroll
  for (int t = 0; t < 8; ++t) O[t] = (v8f){};

  float id8[8];
#pragma unroll
  for (int j = 0; j < 8; ++j) id8[j] = input_dot[(size_t)b * CLD + r0 + j + half * 8];

  for (int it = 0; it < CLM / 32; ++it) {
    int m0 = it * 32, cb = it & 1;
    wait_async0();      // this wave's async loads for buf[cb] have landed
    __syncthreads();    // all waves landed theirs AND finished reading buf[cb^1]
    if (it + 1 < CLM / 32) issue_chunk(cb ^ 1, m0 + 32);  // prefetch next chunk

    v8f S0 = {}, S1 = {};
#pragma unroll
    for (int c = 0; c < 4; ++c) {
      S0 = wmma_f16(xa[c], load_B_lds(&memS[cb][ln][c * 32 + half * 16]), S0);
      S1 = wmma_f16(xa[c], load_B_lds(&memS[cb][16 + ln][c * 32 + half * 16]), S1);
    }
    float md0 = mdotb[m0 + ln], md1 = mdotb[m0 + 16 + ln];
    float cmax[8];
#pragma unroll
    for (int j = 0; j < 8; ++j) {
      S0[j] += id8[j] + md0;
      S1[j] += id8[j] + md1;
      float v = fmaxf(S0[j], S1[j]);
      v = fmaxf(v, __shfl_xor(v, 1, 16));
      v = fmaxf(v, __shfl_xor(v, 2, 16));
      v = fmaxf(v, __shfl_xor(v, 4, 16));
      v = fmaxf(v, __shfl_xor(v, 8, 16));
      cmax[j] = v;
    }
    float alpha[8];
#pragma unroll
    for (int j = 0; j < 8; ++j) {
      float mnew = fmaxf(mstat[j], cmax[j]);
      alpha[j] = __expf(mstat[j] - mnew);
      mstat[j] = mnew;
      float p0 = __expf(S0[j] - mnew), p1 = __expf(S1[j] - mnew);
      float rs = p0 + p1;
      rs += __shfl_xor(rs, 1, 16);
      rs += __shfl_xor(rs, 2, 16);
      rs += __shfl_xor(rs, 4, 16);
      rs += __shfl_xor(rs, 8, 16);
      lstat[j] = lstat[j] * alpha[j] + rs;
      Pbuf[wave][j + half * 8][ln]      = (_Float16)p0;   // (row, local col)
      Pbuf[wave][j + half * 8][16 + ln] = (_Float16)p1;
    }
    asm volatile("s_wait_dscnt 0" ::: "memory");
    // re-layout P: C-fragment -> A-fragment (16x32)
    v16h pf;
#pragma unroll
    for (int j = 0; j < 8; ++j) pf[j] = Pbuf[wave][ln][j + 16 * half];
#pragma unroll
    for (int j = 0; j < 8; ++j) pf[8 + j] = Pbuf[wave][ln][8 + j + 16 * half];

#pragma unroll
    for (int t = 0; t < 8; ++t) {
#pragma unroll
      for (int j = 0; j < 8; ++j) O[t][j] *= alpha[j];
      O[t] = wmma_f16(pf, load_B_lds(&mem2S[cb][t * 16 + ln][half * 16]), O[t]);
    }
  }

  // finalize: o1 = O / l, rowmax = running max
#pragma unroll
  for (int t = 0; t < 8; ++t) {
    int colg = t * 16 + ln;
#pragma unroll
    for (int j = 0; j < 8; ++j) {
      int rg = r0 + j + half * 8;
      o1h[((size_t)b * CLD + rg) * CH + colg] = (_Float16)(O[t][j] / lstat[j]);
    }
  }
  if (ln == 0) {
#pragma unroll
    for (int j = 0; j < 8; ++j) rowmax[(size_t)b * CLD + r0 + j + half * 8] = mstat[j];
  }
}

// ---------------- weight_two / output_two (3-phase, deterministic) ----------------
__global__ __launch_bounds__(128)
void wtwo_stats_kernel(const float* __restrict__ rowmax, float* __restrict__ stats) {
  __shared__ float red[128];
  int b = blockIdx.x, t = threadIdx.x;
  const float* rm = rowmax + (size_t)b * CLD;
  float vmax = -3.0e38f;
  for (int i = t; i < CLD; i += 128) vmax = fmaxf(vmax, rm[i]);
  red[t] = vmax; __syncthreads();
  for (int s = 64; s > 0; s >>= 1) { if (t < s) red[t] = fmaxf(red[t], red[t + s]); __syncthreads(); }
  float mx = red[0];
  __syncthreads();
  float vs = 0.f;
  for (int i = t; i < CLD; i += 128) vs += __expf(rm[i] - mx);
  red[t] = vs; __syncthreads();
  for (int s = 64; s > 0; s >>= 1) { if (t < s) red[t] += red[t + s]; __syncthreads(); }
  if (t == 0) { stats[b * 2 + 0] = mx; stats[b * 2 + 1] = 1.f / red[0]; }
}

// grid (B, LD/128): each block handles 128 rows; softmax weights staged once in LDS.
__global__ __launch_bounds__(128)
void wtwo_acc_kernel(const float* __restrict__ rowmax, const float* __restrict__ stats,
                     const _Float16* __restrict__ inp2h, float* __restrict__ o2part) {
  __shared__ float wbuf[128];
  int b = blockIdx.x, chunk = blockIdx.y, t = threadIdx.x;
  int i0 = chunk * 128;
  float mx = stats[b * 2 + 0], inv = stats[b * 2 + 1];
  wbuf[t] = __expf(rowmax[(size_t)b * CLD + i0 + t] - mx) * inv;
  __syncthreads();
  float acc = 0.f;
  const _Float16* ib = inp2h + ((size_t)b * CLD + i0) * CH;
#pragma unroll 4
  for (int i = 0; i < 128; ++i)
    acc += wbuf[i] * (float)ib[(size_t)i * CH + t];
  o2part[((size_t)b * (CLD / 128) + chunk) * CH + t] = acc;
}

__global__ __launch_bounds__(128)
void wtwo_reduce_kernel(const float* __restrict__ o2part, float* __restrict__ o2) {
  int b = blockIdx.x, t = threadIdx.x;
  float acc = 0.f;
#pragma unroll
  for (int c = 0; c < CLD / 128; ++c) acc += o2part[((size_t)b * (CLD / 128) + c) * CH + t];
  o2[b * CH + t] = acc;  // fixed reduction order -> deterministic
}

// ---------------- final: out = cat @ W_out^T + b_out  (4 K=128 terms) ----------------
__global__ __launch_bounds__(256)
void final_kernel(const _Float16* __restrict__ inp2h, const _Float16* __restrict__ o1h,
                  const float* __restrict__ o2, const _Float16* __restrict__ Wch,
                  const float* __restrict__ b_out, float* __restrict__ out) {
  int wave = threadIdx.x >> 5, lane = threadIdx.x & 31;
  int half = lane >> 4, ln = lane & 15;
  int b = blockIdx.x;
  int r0 = blockIdx.y * 128 + wave * 16;
  const _Float16* ib = inp2h + ((size_t)b * CLD + r0) * CH;
  const _Float16* ob = o1h  + ((size_t)b * CLD + r0) * CH;
  v16h iF[4], oF[4], pF[4], tF[4];
#pragma unroll
  for (int c = 0; c < 4; ++c) {
    iF[c] = load_A16x32(ib + c * 32, CH);
    oF[c] = load_A16x32(ob + c * 32, CH);
    pF[c] = iF[c] * oF[c];  // inp2 * output_one, elementwise on fragments
  }
  const float* o2b = o2 + b * CH;
#pragma unroll
  for (int c = 0; c < 4; ++c) {  // o2 broadcast along rows -> A-fragment of (o2 * o1)
    v16h s;
#pragma unroll
    for (int j = 0; j < 8; ++j) s[j] = (_Float16)o2b[c * 32 + j + 16 * half];
#pragma unroll
    for (int j = 0; j < 8; ++j) s[8 + j] = (_Float16)o2b[c * 32 + 8 + j + 16 * half];
    tF[c] = s * oF[c];
  }
#pragma unroll
  for (int nt = 0; nt < 8; ++nt) {
    v8f acc = {};
#pragma unroll
    for (int c = 0; c < 4; ++c) {
      acc = wmma_f16(iF[c], load_BT32x16(Wch + 0 * CH * CH + (size_t)(nt * 16) * CH + c * 32, CH), acc);
      acc = wmma_f16(oF[c], load_BT32x16(Wch + 1 * CH * CH + (size_t)(nt * 16) * CH + c * 32, CH), acc);
      acc = wmma_f16(pF[c], load_BT32x16(Wch + 2 * CH * CH + (size_t)(nt * 16) * CH + c * 32, CH), acc);
      acc = wmma_f16(tF[c], load_BT32x16(Wch + 3 * CH * CH + (size_t)(nt * 16) * CH + c * 32, CH), acc);
    }
    int colg = nt * 16 + ln;
    float bias = b_out[colg];
#pragma unroll
    for (int j = 0; j < 8; ++j) {
      int rg = r0 + j + half * 8;
      out[((size_t)b * CLD + rg) * CH + colg] = acc[j] + bias;
    }
  }
}

// ---------------- host launcher ----------------
extern "C" void kernel_launch(void* const* d_in, const int* in_sizes, int n_in,
                              void* d_out, int out_size, void* d_ws, size_t ws_size,
                              hipStream_t stream) {
  (void)in_sizes; (void)n_in; (void)out_size; (void)ws_size;
  const float* x     = (const float*)d_in[0];
  const float* mem   = (const float*)d_in[1];
  const float* w_i1  = (const float*)d_in[2];
  const float* w_m1  = (const float*)d_in[3];
  const float* W_i2  = (const float*)d_in[4];
  const float* b_i2  = (const float*)d_in[5];
  const float* W_m2  = (const float*)d_in[6];
  const float* b_m2  = (const float*)d_in[7];
  const float* W_out = (const float*)d_in[8];
  const float* b_out = (const float*)d_in[9];
  float* out = (float*)d_out;

  char* p = (char*)d_ws;
  auto carve = [&](size_t bytes) -> void* {
    void* r = (void*)p;
    p += (bytes + 255) & ~(size_t)255;
    return r;
  };
  _Float16* xh     = (_Float16*)carve((size_t)CB * CLD * CH * 2);
  _Float16* inp2h  = (_Float16*)carve((size_t)CB * CLD * CH * 2);
  _Float16* o1h    = (_Float16*)carve((size_t)CB * CLD * CH * 2);
  _Float16* memh   = (_Float16*)carve((size_t)CB * CLM * CH * 2);
  _Float16* mem2T  = (_Float16*)carve((size_t)CB * CH * CLM * 2);
  float*    idot   = (float*)carve((size_t)CB * CLD * 4);
  float*    mdot   = (float*)carve((size_t)CB * CLM * 4);
  float*    rmax   = (float*)carve((size_t)CB * CLD * 4);
  float*    stats  = (float*)carve((size_t)CB * 2 * 4);
  float*    o2part = (float*)carve((size_t)CB * (CLD / 128) * CH * 4);
  float*    o2     = (float*)carve((size_t)CB * CH * 4);
  _Float16* Wi2h   = (_Float16*)carve((size_t)CH * CH * 2);
  _Float16* Wch    = (_Float16*)carve((size_t)4 * CH * CH * 2);

  prep_x_kernel<<<CB * CLD, 128, 0, stream>>>(x, w_i1, xh, idot);
  prep_mem_kernel<<<CB * CLM, 128, 0, stream>>>(mem, w_m1, W_m2, b_m2, memh, mem2T, mdot);
  prep_w_kernel<<<(4 * CH * CH + 255) / 256, 256, 0, stream>>>(W_i2, W_out, Wi2h, Wch);
  attn_kernel<<<dim3(CB, CLD / 128), 256, 0, stream>>>(xh, memh, mem2T, Wi2h, b_i2, idot,
                                                       mdot, inp2h, o1h, rmax);
  wtwo_stats_kernel<<<CB, 128, 0, stream>>>(rmax, stats);
  wtwo_acc_kernel<<<dim3(CB, CLD / 128), 128, 0, stream>>>(rmax, stats, inp2h, o2part);
  wtwo_reduce_kernel<<<CB, 128, 0, stream>>>(o2part, o2);
  final_kernel<<<dim3(CB, CLD / 128), 256, 0, stream>>>(inp2h, o1h, o2, Wch, b_out, out);
}
